// SharedAttention_73890617361055
// MI455X (gfx1250) — matrix operand order repaired and checked by reading, not compiled, not run on hardware
//
#include <hip/hip_runtime.h>
#include <hip/hip_bf16.h>

typedef __attribute__((ext_vector_type(16))) _Float16 v16h;
typedef __attribute__((ext_vector_type(8)))  _Float16 v8h;
typedef __attribute__((ext_vector_type(8)))  float    v8f;

union V16U { v16h v; v8h h[2]; };

// ---------------------------------------------------------------------------
// WMMA helpers (CDNA5: V_WMMA_F32_16X16X32_F16, wave32)
// ---------------------------------------------------------------------------
__device__ inline v8f wmma_f16(v16h a, v16h b, v8f c) {
  return __builtin_amdgcn_wmma_f32_16x16x32_f16(
      /*neg_a=*/false, a, /*neg_b=*/false, b,
      /*c_mod=*/(short)0, c, /*reuse_a=*/false, /*reuse_b=*/false);
}

// A fragment: 16x32 (MxK) f16 tile from row-major matrix, leading dim ld.
// Lane L (L<16) holds row M=L with K = {k0..k0+15};
// lane L+16 holds row M=L with K = {k0+16..k0+31}.
__device__ inline v16h load_a_frag(const _Float16* __restrict__ base,
                                   int row0, int k0, int ld, int lane) {
  const int row = row0 + (lane & 15);
  const int ko  = (lane >> 4) * 16;
  const _Float16* p = base + row * ld + k0 + ko;
  return *(const v16h*)(p);
}

// B fragment: 32x16 (KxN) where B[k][n] = W[n][k], W row-major leading dim ld.
// Lane n (n<16) holds B[k0..k0+15][n0+n] contiguously; lane n+16 holds K+16.
__device__ inline v16h load_b_frag(const _Float16* __restrict__ W,
                                   int n0, int k0, int ld, int lane) {
  const int col = n0 + (lane & 15);
  const int ko  = (lane >> 4) * 16;
  return *(const v16h*)(W + col * ld + k0 + ko);
}

// ---------------------------------------------------------------------------
// Prep kernels
// ---------------------------------------------------------------------------
__global__ void cvt_f16_kernel(const float* __restrict__ in,
                               _Float16* __restrict__ out, int n) {
  int idx = blockIdx.x * 256 + threadIdx.x;
  if (idx < n) out[idx] = (_Float16)in[idx];
}

// qkv_w[o,i] = sum_t 0.5*(coeffs[0,t]+coeffs[1,t]) * templates[t,o,i]
__global__ void build_qkvw_kernel(const float* __restrict__ T,
                                  const float* __restrict__ coeffs,
                                  _Float16* __restrict__ Wh) {
  int idx = blockIdx.x * 256 + threadIdx.x;
  if (idx >= 2304 * 768) return;
  float acc = 0.f;
#pragma unroll
  for (int t = 0; t < 16; ++t) {
    float c = 0.5f * (coeffs[t] + coeffs[16 + t]);
    acc += c * T[(size_t)t * (2304u * 768u) + idx];
  }
  Wh[idx] = (_Float16)acc;
}

// ---------------------------------------------------------------------------
// QKV GEMM: (4096x768) x (768x2304) ; epilogue scatters Q (scaled), K, V^T
// Wave computes a 64x64 output tile. 2304 waves total (576 blocks of 128).
// ---------------------------------------------------------------------------
__global__ __launch_bounds__(128) void qkv_gemm_kernel(
    const _Float16* __restrict__ X, const _Float16* __restrict__ Wq,
    const float* __restrict__ bias, _Float16* __restrict__ Q,
    _Float16* __restrict__ K, _Float16* __restrict__ VT) {
  const int lane = threadIdx.x & 31;
  const int wg   = blockIdx.x * 4 + (threadIdx.x >> 5);
  const int mt = wg / 36, nt = wg - 36 * (wg / 36);
  const int m0 = mt * 64, n0 = nt * 64;

  v8f acc[4][4];
#pragma unroll
  for (int i = 0; i < 4; ++i)
#pragma unroll
    for (int j = 0; j < 4; ++j) acc[i][j] = {};

  for (int k0 = 0; k0 < 768; k0 += 32) {
    if (k0 + 32 < 768)
      __builtin_prefetch(X + (m0 + (lane & 15)) * 768 + k0 + 32, 0, 1);
    v16h a[4];
#pragma unroll
    for (int mi = 0; mi < 4; ++mi)
      a[mi] = load_a_frag(X, m0 + mi * 16, k0, 768, lane);
#pragma unroll
    for (int ni = 0; ni < 4; ++ni) {
      v16h b = load_b_frag(Wq, n0 + ni * 16, k0, 768, lane);
#pragma unroll
      for (int mi = 0; mi < 4; ++mi) acc[mi][ni] = wmma_f16(a[mi], b, acc[mi][ni]);
    }
  }

  const float qscale = 0.125f;  // hd^-0.5 folded into Q
#pragma unroll
  for (int ni = 0; ni < 4; ++ni) {
#pragma unroll
    for (int mi = 0; mi < 4; ++mi) {
#pragma unroll
      for (int j = 0; j < 8; ++j) {
        int row = m0 + mi * 16 + j + (lane >> 4) * 8;  // 0..4095
        int col = n0 + ni * 16 + (lane & 15);          // 0..2303
        float v = acc[mi][ni][j] + bias[col];
        int sect = col / 768;
        int within = col - sect * 768;
        int h = within >> 6, d = within & 63;
        if (sect == 0)
          Q[((size_t)h * 4096 + row) * 64 + d] = (_Float16)(v * qscale);
        else if (sect == 1)
          K[((size_t)h * 4096 + row) * 64 + d] = (_Float16)v;
        else
          VT[((size_t)h * 64 + d) * 4096 + row] = (_Float16)v;
      }
    }
  }
}

// ---------------------------------------------------------------------------
// Flash attention: one wave handles 16 query rows of one head, streaming over
// all 4096 kv positions in chunks of 32. Online softmax; P goes through LDS
// to convert WMMA C-layout -> A-layout (s_wait_dscnt for the RAW hazard).
// 12 heads * 256 qblocks = 3072 waves (768 blocks of 128).
// ---------------------------------------------------------------------------
__global__ __launch_bounds__(128) void flash_attn_kernel(
    const _Float16* __restrict__ Q, const _Float16* __restrict__ K,
    const _Float16* __restrict__ VT, _Float16* __restrict__ O) {
  __shared__ __align__(32) _Float16 Pbuf[4][16 * 32];
  const int lane = threadIdx.x & 31;
  const int wid  = threadIdx.x >> 5;
  const int wg = blockIdx.x * 4 + wid;
  const int h  = wg >> 8;    // 0..11
  const int qb = wg & 255;   // 0..255

  const _Float16* Qh = Q + (size_t)h * 4096 * 64;
  const _Float16* Kh = K + (size_t)h * 4096 * 64;
  const _Float16* Vh = VT + (size_t)h * 64 * 4096;

  const v16h aq0 = load_a_frag(Qh, qb * 16, 0, 64, lane);
  const v16h aq1 = load_a_frag(Qh, qb * 16, 32, 64, lane);

  v8f o[4];
#pragma unroll
  for (int t = 0; t < 4; ++t) o[t] = {};
  float mrow[8], lrow[8];
#pragma unroll
  for (int j = 0; j < 8; ++j) { mrow[j] = -3.0e38f; lrow[j] = 0.f; }

  _Float16* pb = Pbuf[wid];

  for (int kc = 0; kc < 4096; kc += 32) {
    if (kc + 32 < 4096)
      __builtin_prefetch(Kh + (kc + 32 + (lane & 15)) * 64, 0, 1);

    // S = (q*scale) . k^T   : two 16x16 tiles along kv, K-dim = hd = 64
    v8f s0 = {}, s1 = {};
    {
      v16h b00 = load_b_frag(Kh, kc,      0,  64, lane);
      v16h b01 = load_b_frag(Kh, kc,      32, 64, lane);
      v16h b10 = load_b_frag(Kh, kc + 16, 0,  64, lane);
      v16h b11 = load_b_frag(Kh, kc + 16, 32, 64, lane);
      s0 = wmma_f16(aq0, b00, s0);
      s0 = wmma_f16(aq1, b01, s0);
      s1 = wmma_f16(aq0, b10, s1);
      s1 = wmma_f16(aq1, b11, s1);
    }

    // Online softmax. Row m of the C tile lives in VGPR j across a 16-lane
    // half (lanes 0-15: m=j, lanes 16-31: m=8+j).
#pragma unroll
    for (int j = 0; j < 8; ++j) {
      float t = fmaxf(s0[j], s1[j]);
#pragma unroll
      for (int m = 1; m < 16; m <<= 1) t = fmaxf(t, __shfl_xor(t, m, 32));
      float mnew  = fmaxf(mrow[j], t);
      float alpha = __expf(mrow[j] - mnew);
      float p0 = __expf(s0[j] - mnew);
      float p1 = __expf(s1[j] - mnew);
      float rs = p0 + p1;
#pragma unroll
      for (int m = 1; m < 16; m <<= 1) rs += __shfl_xor(rs, m, 32);
      lrow[j] = lrow[j] * alpha + rs;
      mrow[j] = mnew;
#pragma unroll
      for (int t4 = 0; t4 < 4; ++t4) o[t4][j] *= alpha;
      // Stage P (f16) to LDS in [row][kv] order for the A-layout reload.
      int r  = j + (lane >> 4) * 8;
      int c0 = lane & 15;
      pb[r * 32 + c0]      = (_Float16)p0;
      pb[r * 32 + 16 + c0] = (_Float16)p1;
    }

    asm volatile("s_wait_dscnt 0" ::: "memory");  // LDS RAW: stores -> loads

    // Reload P as a 16x32 A fragment (ds_load_b128 pairs).
    v16h pa;
    {
      int row = lane & 15;
      int ko  = (lane >> 4) * 16;
      pa = *(const v16h*)(pb + row * 32 + ko);
    }

    // O += P . V  (V^T layout makes B fragments contiguous along kv)
#pragma unroll
    for (int t4 = 0; t4 < 4; ++t4) {
      const _Float16* vp =
          Vh + ((size_t)(t4 * 16 + (lane & 15))) * 4096 + kc + (lane >> 4) * 16;
      v16h bv = *(const v16h*)vp;
      o[t4] = wmma_f16(pa, bv, o[t4]);
    }
  }

  // Normalize and write attn output (row-major 4096x768, f16)
#pragma unroll
  for (int j = 0; j < 8; ++j) {
    float inv = 1.0f / lrow[j];
    int row = qb * 16 + j + (lane >> 4) * 8;
#pragma unroll
    for (int t4 = 0; t4 < 4; ++t4) {
      int col = h * 64 + t4 * 16 + (lane & 15);
      O[(size_t)row * 768 + col] = (_Float16)(o[t4][j] * inv);
    }
  }
}

// ---------------------------------------------------------------------------
// Output projection: (4096x768) x (768x768)^T + b -> fp32. 768 waves.
// ---------------------------------------------------------------------------
__global__ __launch_bounds__(128) void proj_gemm_kernel(
    const _Float16* __restrict__ A, const _Float16* __restrict__ W,
    const float* __restrict__ bias, float* __restrict__ Out) {
  const int lane = threadIdx.x & 31;
  const int wg   = blockIdx.x * 4 + (threadIdx.x >> 5);
  const int mt = wg / 12, nt = wg - 12 * (wg / 12);
  const int m0 = mt * 64, n0 = nt * 64;

  v8f acc[4][4];
#pragma unroll
  for (int i = 0; i < 4; ++i)
#pragma unroll
    for (int j = 0; j < 4; ++j) acc[i][j] = {};

  for (int k0 = 0; k0 < 768; k0 += 32) {
    if (k0 + 32 < 768)
      __builtin_prefetch(A + (m0 + (lane & 15)) * 768 + k0 + 32, 0, 1);
    v16h a[4];
#pragma unroll
    for (int mi = 0; mi < 4; ++mi)
      a[mi] = load_a_frag(A, m0 + mi * 16, k0, 768, lane);
#pragma unroll
    for (int ni = 0; ni < 4; ++ni) {
      v16h b = load_b_frag(W, n0 + ni * 16, k0, 768, lane);
#pragma unroll
      for (int mi = 0; mi < 4; ++mi) acc[mi][ni] = wmma_f16(a[mi], b, acc[mi][ni]);
    }
  }

#pragma unroll
  for (int ni = 0; ni < 4; ++ni) {
#pragma unroll
    for (int mi = 0; mi < 4; ++mi) {
#pragma unroll
      for (int j = 0; j < 8; ++j) {
        int row = m0 + mi * 16 + j + (lane >> 4) * 8;
        int col = n0 + ni * 16 + (lane & 15);
        Out[(size_t)row * 768 + col] = acc[mi][ni][j] + bias[col];
      }
    }
  }
}

// ---------------------------------------------------------------------------
// Launch
// ---------------------------------------------------------------------------
extern "C" void kernel_launch(void* const* d_in, const int* in_sizes, int n_in,
                              void* d_out, int out_size, void* d_ws, size_t ws_size,
                              hipStream_t stream) {
  const float* x         = (const float*)d_in[0];  // (1,4096,768)
  const float* templates = (const float*)d_in[1];  // (16,2304,768)
  const float* coeffs    = (const float*)d_in[2];  // (2,16,1)
  const float* qkv_bias  = (const float*)d_in[3];  // (2304,)
  const float* proj_w    = (const float*)d_in[4];  // (768,768)
  const float* proj_b    = (const float*)d_in[5];  // (768,)
  float* out = (float*)d_out;                      // (1,4096,768) fp32

  char* ws = (char*)d_ws;
  _Float16* x_h    = (_Float16*)(ws + 0);         // 4096*768   f16 =  6.0 MB
  _Float16* qkvw_h = (_Float16*)(ws + 6291456);   // 2304*768   f16 =  3.4 MB
  _Float16* projw_h= (_Float16*)(ws + 9830400);   // 768*768    f16 =  1.1 MB
  _Float16* Q      = (_Float16*)(ws + 11010048);  // 12*4096*64 f16 =  6.0 MB
  _Float16* Kk     = (_Float16*)(ws + 17301504);  // 12*4096*64 f16 =  6.0 MB
  _Float16* VT     = (_Float16*)(ws + 23592960);  // 12*64*4096 f16 =  6.0 MB
  _Float16* AO     = (_Float16*)(ws + 29884416);  // 4096*768   f16 =  6.0 MB
  // total ws use: 36,175,872 bytes

  cvt_f16_kernel<<<(4096 * 768 + 255) / 256, 256, 0, stream>>>(x, x_h, 4096 * 768);
  cvt_f16_kernel<<<(768 * 768 + 255) / 256, 256, 0, stream>>>(proj_w, projw_h, 768 * 768);
  build_qkvw_kernel<<<(2304 * 768 + 255) / 256, 256, 0, stream>>>(templates, coeffs, qkvw_h);

  qkv_gemm_kernel<<<576, 128, 0, stream>>>(x_h, qkvw_h, qkv_bias, Q, Kk, VT);
  flash_attn_kernel<<<768, 128, 0, stream>>>(Q, Kk, VT, AO);
  proj_gemm_kernel<<<192, 128, 0, stream>>>(AO, projw_h, proj_b, out);
}
